// Model_53979148976153
// MI455X (gfx1250) — matrix operand order repaired and checked by reading, not compiled, not run on hardware
//
#include <hip/hip_runtime.h>
#include <hip/hip_bf16.h>
#include <math.h>
#include <stdint.h>

#define SRC_LEN 2048
#define TRG_LEN 512
#define BSZ     64
#define ENC_IN  8
#define HDIM    256
#define G3      (3 * HDIM)   // 768
#define H2      (2 * HDIM)   // 512
#define KW_DEC  513          // 2H + 1

typedef __attribute__((ext_vector_type(16))) _Float16 v16h;
typedef __attribute__((ext_vector_type(8)))  float    v8f;
typedef unsigned int u32x4 __attribute__((ext_vector_type(4)));
typedef int          i32x4 __attribute__((ext_vector_type(4)));
typedef int          i32x8 __attribute__((ext_vector_type(8)));

#define WMMA_F16(a, b, c) \
  __builtin_amdgcn_wmma_f32_16x16x32_f16(false, (a), false, (b), (short)0, (c), false, false)

// ---------------------------------------------------------------------------
// Fragment loaders (wave32 WMMA layouts per CDNA5 ISA 7.12.2)
// A: 16x32 f16 tile, row-major src. lane<16: rows M=lane, K = {0..7,16..23};
//    lane>=16: same rows, K = {8..15,24..31}.
// B: 32x16 tile loaded from B^T stored row-major (N x K): lane%16 = N,
//    lane<16: K = {0..7,16..23}; lane>=16: K = {8..15,24..31} (matches A).
// ---------------------------------------------------------------------------
template <typename T>
__device__ __forceinline__ v16h load_a_rm(const T* A, int lda, int lane) {
  const int m  = lane & 15;
  const int kb = (lane >> 4) << 3;
  const T* p = A + m * lda + kb;
  v16h a;
#pragma unroll
  for (int i = 0; i < 8; ++i) {
    a[i]     = (_Float16)p[i];
    a[8 + i] = (_Float16)p[16 + i];
  }
  return a;
}

__device__ __forceinline__ v16h load_bt_rm(const _Float16* BT, int ldb, int lane) {
  const int n  = lane & 15;
  const int kb = (lane >> 4) << 3;
  const _Float16* p = BT + n * ldb + kb;
  v16h b;
#pragma unroll
  for (int i = 0; i < 8; ++i) { b[i] = p[i]; b[8 + i] = p[16 + i]; }
  return b;
}

__device__ __forceinline__ float sigf(float x) { return 1.0f / (1.0f + __expf(-x)); }

// ---------------------------------------------------------------------------
// f32 -> f16 weight conversion (plain, and dec_Wih repack dropping col 0 so
// B-fragments stay 16B aligned: dst[n*512 + k] = src[n*513 + 1 + k])
// ---------------------------------------------------------------------------
__global__ void k_cvt(const float* __restrict__ s, _Float16* __restrict__ d, int n) {
  int i = blockIdx.x * 256 + threadIdx.x;
  if (i < n) d[i] = (_Float16)s[i];
}

__global__ void k_cvt_dwih(const float* __restrict__ s, _Float16* __restrict__ d) {
  int i = blockIdx.x * 256 + threadIdx.x;      // over 768*512
  int n = i >> 9, k = i & 511;
  d[i] = (_Float16)s[n * KW_DEC + 1 + k];
}

// ---------------------------------------------------------------------------
// Bidirectional GRU encoder: one persistent workgroup per direction.
// h kept in LDS (f16). Two-phase step to avoid read/write hazards without
// preloading A-fragments.
// ---------------------------------------------------------------------------
__global__ __launch_bounds__(1024) void k_encoder(
    const float* __restrict__ x,
    const float* __restrict__ WihF, const float* __restrict__ bihF,
    const float* __restrict__ bhhF,
    const float* __restrict__ WihB, const float* __restrict__ bihB,
    const float* __restrict__ bhhB,
    const _Float16* __restrict__ WhhF16, const _Float16* __restrict__ WhhB16,
    _Float16* __restrict__ enc16) {
  __shared__ _Float16 hs[BSZ * HDIM];   // 32 KB hidden state
  __shared__ float    xs[BSZ * ENC_IN]; // 2 KB current input

  const int dir = blockIdx.x;
  const float*    Wih   = dir ? WihB : WihF;
  const float*    bih   = dir ? bihB : bihF;
  const float*    bhh   = dir ? bhhB : bhhF;
  const _Float16* Whh16 = dir ? WhhB16 : WhhF16;

  const int tid = threadIdx.x, lane = tid & 31, wave = tid >> 5;
  const int mtile = wave >> 3;          // 0..3 (rows mtile*16..)
  const int j0    = (wave & 7) * 2;     // this wave's 2 j-tiles
  const int nloc  = lane & 15;
  const int mb    = mtile * 16 + ((lane >> 4) << 3);

  for (int i = tid; i < BSZ * HDIM; i += 1024) hs[i] = (_Float16)0.f;

  for (int step = 0; step < SRC_LEN; ++step) {
    const int t = dir ? (SRC_LEN - 1 - step) : step;

    __syncthreads();  // previous epilogue's hs/xs use complete
    if (tid < BSZ * ENC_IN) xs[tid] = x[(size_t)t * BSZ * ENC_IN + tid];

    v8f aR[2] = {{}, {}}, aZ[2] = {{}, {}}, aN[2] = {{}, {}};
#pragma unroll 1
    for (int k = 0; k < 8; ++k) {
      v16h a = load_a_rm(hs + (mtile * 16) * HDIM + k * 32, HDIM, lane);
#pragma unroll
      for (int jt = 0; jt < 2; ++jt) {
        const int j = j0 + jt;
        v16h b;
        b = load_bt_rm(Whh16 + (0 * HDIM + j * 16) * HDIM + k * 32, HDIM, lane);
        aR[jt] = WMMA_F16(a, b, aR[jt]);
        b = load_bt_rm(Whh16 + (1 * HDIM + j * 16) * HDIM + k * 32, HDIM, lane);
        aZ[jt] = WMMA_F16(a, b, aZ[jt]);
        b = load_bt_rm(Whh16 + (2 * HDIM + j * 16) * HDIM + k * 32, HDIM, lane);
        aN[jt] = WMMA_F16(a, b, aN[jt]);
      }
    }
    __syncthreads();  // all hs reads done; xs staged

#pragma unroll
    for (int jt = 0; jt < 2; ++jt) {
      const int j   = j0 + jt;
      const int col = j * 16 + nloc;
      float wr[ENC_IN], wz[ENC_IN], wn[ENC_IN];
#pragma unroll
      for (int k = 0; k < ENC_IN; ++k) {
        wr[k] = Wih[(0 * HDIM + col) * ENC_IN + k];
        wz[k] = Wih[(1 * HDIM + col) * ENC_IN + k];
        wn[k] = Wih[(2 * HDIM + col) * ENC_IN + k];
      }
      const float br_ = bih[col] + bhh[col];
      const float bz_ = bih[HDIM + col] + bhh[HDIM + col];
      const float bxn = bih[2 * HDIM + col];
      const float bhn = bhh[2 * HDIM + col];
#pragma unroll
      for (int r = 0; r < 8; ++r) {
        const int m = mb + r;
        float gr = aR[jt][r] + br_;
        float gz = aZ[jt][r] + bz_;
        float gxn = bxn;
#pragma unroll
        for (int k = 0; k < ENC_IN; ++k) {
          const float xv = xs[m * ENC_IN + k];
          gr  += xv * wr[k];
          gz  += xv * wz[k];
          gxn += xv * wn[k];
        }
        const float hn_  = aN[jt][r] + bhn;
        const float rg   = sigf(gr);
        const float zg   = sigf(gz);
        const float ng   = tanhf(gxn + rg * hn_);
        const float hold = (float)hs[m * HDIM + col];
        const float hnew = (1.f - zg) * ng + zg * hold;
        hs[m * HDIM + col] = (_Float16)hnew;  // own region only
        enc16[((size_t)m * SRC_LEN + t) * H2 + dir * HDIM + col] = (_Float16)hnew;
      }
    }
  }
}

// ---------------------------------------------------------------------------
// E = enc @ attn_We^T  (M = B*S = 131072, N = 256, K = 512).
// One block (16 waves) per 16-row M-tile: the 16x512 f16 A strip (contiguous
// 16 KB) is DMA'd into LDS once via the Tensor Data Mover, then each wave
// computes one of the 16 N-tiles reading A from LDS (16x less L2 traffic).
// ---------------------------------------------------------------------------
__global__ __launch_bounds__(512) void k_gemm_E(
    const _Float16* __restrict__ enc16, const _Float16* __restrict__ attnW16,
    _Float16* __restrict__ E16) {
  __shared__ _Float16 As[16 * H2];  // 16 KB A tile
  const int lane = threadIdx.x & 31, wave = threadIdx.x >> 5;
  const int mt = blockIdx.x;        // 8192 m-tiles
  const int nt = wave;              // 16 n-tiles

  if (threadIdx.x < 32) {           // wave 0 issues the TDM transfer
    const unsigned long long ga =
        (unsigned long long)(uintptr_t)(enc16 + (size_t)mt * 16 * H2);
    const unsigned int lds = (unsigned int)(uintptr_t)As;
    // D# group 0: count=1 | lds_addr | global_addr[56:0] | type=2
    u32x4 g0;
    g0[0] = 1u;
    g0[1] = lds;
    g0[2] = (unsigned int)(ga & 0xFFFFFFFFu);
    g0[3] = (unsigned int)((ga >> 32) & 0x01FFFFFFu) | (2u << 30);
    // D# group 1: data_size=3 (8B), 1-D tile of 2048 units (= 16 KB)
    i32x8 g1;
    g1[0] = 3 << 16;                 // data_size=3, no mask/barrier/iterate
    g1[1] = (2048 & 0xFFFF) << 16;   // tensor_dim0[15:0]
    g1[2] = (2048 >> 16) | (1 << 16);// tensor_dim0[31:16] | tensor_dim1=1
    g1[3] = 2048 << 16;              // tile_dim0 = 2048
    g1[4] = 1;                       // tile_dim1 = 1, tile_dim2 = 0
    g1[5] = 2048;                    // tensor_dim0_stride[31:0]
    g1[6] = (2048 & 0xFFFF) << 16;   // tensor_dim1_stride[15:0]
    g1[7] = 0;
    i32x4 gz = {0, 0, 0, 0};
#if __clang_major__ >= 23
    i32x8 gz8 = {0, 0, 0, 0, 0, 0, 0, 0};
    __builtin_amdgcn_tensor_load_to_lds(g0, g1, gz, gz, gz8, 0);
#else
    __builtin_amdgcn_tensor_load_to_lds(g0, g1, gz, gz, 0);
#endif
    __builtin_amdgcn_s_wait_tensorcnt((short)0);
  }
  __syncthreads();

  const _Float16* Bbase = attnW16 + (nt * 16) * G3 + HDIM;  // cols 256..767
  v8f acc = {};
#pragma unroll 4
  for (int k = 0; k < 16; ++k) {
    v16h a = load_a_rm(As + k * 32, H2, lane);
    v16h b = load_bt_rm(Bbase + k * 32, G3, lane);
    acc = WMMA_F16(a, b, acc);
  }
  const int n  = nt * 16 + (lane & 15);
  const int m0 = mt * 16 + ((lane >> 4) << 3);
#pragma unroll
  for (int r = 0; r < 8; ++r)
    E16[(size_t)(m0 + r) * HDIM + n] = (_Float16)acc[r];
}

// ---------------------------------------------------------------------------
// hidden = tanh([hf, hb] @ enc_fc_W^T + b) -> decoder h0 (f32 + f16)
// A rows: k<256 -> enc16[b, S-1, k] (fwd final); k>=256 -> enc16[b, 0, k].
// ---------------------------------------------------------------------------
__device__ __forceinline__ v16h load_a_hidden(const _Float16* enc16, int m0,
                                              int ktile, int lane) {
  const int m  = m0 + (lane & 15);
  const int kb = ktile * 32 + ((lane >> 4) << 3);
  const size_t base   = (size_t)m * SRC_LEN * H2;
  const size_t rowoff = (kb < HDIM) ? (size_t)(SRC_LEN - 1) * H2 : 0;
  const _Float16* p = enc16 + base + rowoff + kb;
  v16h a;
#pragma unroll
  for (int i = 0; i < 8; ++i) { a[i] = p[i]; a[8 + i] = p[16 + i]; }
  return a;
}

__global__ __launch_bounds__(1024) void k_enc_hidden(
    const _Float16* __restrict__ enc16, const _Float16* __restrict__ fcW16,
    const float* __restrict__ fcb, float* __restrict__ h32,
    _Float16* __restrict__ h16) {
  const int lane = threadIdx.x & 31, wave = threadIdx.x >> 5;
#pragma unroll 1
  for (int tt = 0; tt < 2; ++tt) {
    const int tile = wave * 2 + tt;
    const int mt = tile >> 4, nt = tile & 15;
    v8f acc = {};
#pragma unroll 4
    for (int k = 0; k < 16; ++k) {
      v16h a = load_a_hidden(enc16, mt * 16, k, lane);
      v16h b = load_bt_rm(fcW16 + (nt * 16) * H2 + k * 32, H2, lane);
      acc = WMMA_F16(a, b, acc);
    }
    const int n  = nt * 16 + (lane & 15);
    const int m0 = mt * 16 + ((lane >> 4) << 3);
#pragma unroll
    for (int r = 0; r < 8; ++r) {
      const float v = tanhf(acc[r] + fcb[n]);
      h32[(m0 + r) * HDIM + n] = v;
      h16[(m0 + r) * HDIM + n] = (_Float16)v;
    }
  }
}

// ---------------------------------------------------------------------------
// q = h @ attn_Wh^T + attn_b  (64x256 @ 256x256), attn_Wh = attn_W[:, 0:256]
// ---------------------------------------------------------------------------
__global__ __launch_bounds__(1024) void k_dec_q(
    const _Float16* __restrict__ h16, const _Float16* __restrict__ attnW16,
    const float* __restrict__ attn_b, float* __restrict__ q) {
  const int lane = threadIdx.x & 31, wave = threadIdx.x >> 5;
#pragma unroll 1
  for (int tt = 0; tt < 2; ++tt) {
    const int tile = wave * 2 + tt;
    const int mt = tile >> 4, nt = tile & 15;
    v8f acc = {};
#pragma unroll 4
    for (int k = 0; k < 8; ++k) {
      v16h a = load_a_rm(h16 + (mt * 16) * HDIM + k * 32, HDIM, lane);
      v16h b = load_bt_rm(attnW16 + (nt * 16) * G3 + k * 32, G3, lane);
      acc = WMMA_F16(a, b, acc);
    }
    const int n  = nt * 16 + (lane & 15);
    const int m0 = mt * 16 + ((lane >> 4) << 3);
#pragma unroll
    for (int r = 0; r < 8; ++r)
      q[(m0 + r) * HDIM + n] = acc[r] + attn_b[n];
  }
}

// ---------------------------------------------------------------------------
// scores[b][s] = sum_h v[h] * tanh(E[b,s,h] + q[b,h]). One wave per (b,s).
// ---------------------------------------------------------------------------
__global__ __launch_bounds__(256) void k_scores(
    const _Float16* __restrict__ E16, const float* __restrict__ q,
    const float* __restrict__ v, float* __restrict__ scores) {
  const int lane = threadIdx.x & 31;
  const int row  = blockIdx.x * 8 + (threadIdx.x >> 5);  // b*S + s
  const int b    = row >> 11;
  const _Float16* ep = E16 + (size_t)row * HDIM + lane * 8;
  const float* qp = q + b * HDIM + lane * 8;
  const float* vp = v + lane * 8;
  float p = 0.f;
#pragma unroll
  for (int i = 0; i < 8; ++i) p += vp[i] * tanhf((float)ep[i] + qp[i]);
#pragma unroll
  for (int off = 16; off; off >>= 1) p += __shfl_xor(p, off);
  if (lane == 0) scores[row] = p;
}

// ---------------------------------------------------------------------------
// Per-batch softmax over S=2048 + weighted = a^T @ enc. One block per batch.
// ---------------------------------------------------------------------------
__global__ __launch_bounds__(1024) void k_attn_apply(
    const float* __restrict__ scores, const _Float16* __restrict__ enc16,
    float* __restrict__ w32, _Float16* __restrict__ w16) {
  __shared__ float aw[SRC_LEN];
  __shared__ float red[32];
  __shared__ float part[1024];
  const int b = blockIdx.x, tid = threadIdx.x;
  const int lane = tid & 31, wave = tid >> 5;

  const float s0 = scores[b * SRC_LEN + tid];
  const float s1 = scores[b * SRC_LEN + 1024 + tid];

  float m = fmaxf(s0, s1);
#pragma unroll
  for (int off = 16; off; off >>= 1) m = fmaxf(m, __shfl_xor(m, off));
  if (lane == 0) red[wave] = m;
  __syncthreads();
  if (tid < 32) {
    float r = red[tid];
#pragma unroll
    for (int off = 16; off; off >>= 1) r = fmaxf(r, __shfl_xor(r, off));
    if (tid == 0) red[0] = r;
  }
  __syncthreads();
  const float M = red[0];
  __syncthreads();

  const float e0 = __expf(s0 - M), e1 = __expf(s1 - M);
  float s = e0 + e1;
#pragma unroll
  for (int off = 16; off; off >>= 1) s += __shfl_xor(s, off);
  if (lane == 0) red[wave] = s;
  __syncthreads();
  if (tid < 32) {
    float r = red[tid];
#pragma unroll
    for (int off = 16; off; off >>= 1) r += __shfl_xor(r, off);
    if (tid == 0) red[0] = r;
  }
  __syncthreads();
  const float inv = 1.0f / red[0];
  aw[tid]        = e0 * inv;
  aw[1024 + tid] = e1 * inv;
  __syncthreads();

  // weighted[b][c] = sum_s a[s] * enc16[b,s,c]; 2 threads per column
  const int c = tid & 511, half = tid >> 9;
  const _Float16* ep = enc16 + ((size_t)b * SRC_LEN + half * 1024) * H2 + c;
  const float* ap = aw + half * 1024;
  float acc = 0.f;
#pragma unroll 4
  for (int ss = 0; ss < 1024; ++ss) acc += ap[ss] * (float)ep[(size_t)ss * H2];
  part[tid] = acc;
  __syncthreads();
  if (half == 0) {
    const float wv = part[c] + part[512 + c];
    w32[b * H2 + c] = wv;
    w16[b * H2 + c] = (_Float16)wv;
  }
}

// ---------------------------------------------------------------------------
// Decoder GRU cell + output projection. rnn_in = [tgt, weighted] (K=513):
// K=512 via WMMA on repacked dec_Wih (cols 1..512, ld=512), tgt*col0 scalar;
// h part K=256 via WMMA. Hidden state double-buffered (hin16 / hout16).
// ---------------------------------------------------------------------------
__global__ __launch_bounds__(1024) void k_dec_gru(
    const float* __restrict__ tgt, const float* __restrict__ w32,
    const _Float16* __restrict__ w16, float* __restrict__ h32,
    const _Float16* __restrict__ hin16, _Float16* __restrict__ hout16,
    const _Float16* __restrict__ dWihP16, const _Float16* __restrict__ dWhh16,
    const float* __restrict__ dWih32, const float* __restrict__ bih,
    const float* __restrict__ bhh, const float* __restrict__ fcW,
    const float* __restrict__ fcb, float* __restrict__ outp) {
  __shared__ _Float16 h2s[BSZ * HDIM];
  const int tid = threadIdx.x, lane = tid & 31, wave = tid >> 5;
  const int mtile = wave >> 3, j0 = (wave & 7) * 2;
  const int nloc = lane & 15;
  const int mb = mtile * 16 + ((lane >> 4) << 3);

#pragma unroll 1
  for (int jt = 0; jt < 2; ++jt) {
    const int j   = j0 + jt;
    const int col = j * 16 + nloc;
    v8f ar = {}, az = {}, anx = {}, anh = {};
#pragma unroll 1
    for (int k = 0; k < 16; ++k) {  // weighted part, K=512
      v16h a = load_a_rm(w16 + (mtile * 16) * H2 + k * 32, H2, lane);
      v16h b;
      b = load_bt_rm(dWihP16 + (0 * HDIM + j * 16) * H2 + k * 32, H2, lane);
      ar  = WMMA_F16(a, b, ar);
      b = load_bt_rm(dWihP16 + (1 * HDIM + j * 16) * H2 + k * 32, H2, lane);
      az  = WMMA_F16(a, b, az);
      b = load_bt_rm(dWihP16 + (2 * HDIM + j * 16) * H2 + k * 32, H2, lane);
      anx = WMMA_F16(a, b, anx);
    }
#pragma unroll 1
    for (int k = 0; k < 8; ++k) {  // hidden part, K=256
      v16h a = load_a_rm(hin16 + (mtile * 16) * HDIM + k * 32, HDIM, lane);
      v16h b;
      b = load_bt_rm(dWhh16 + (0 * HDIM + j * 16) * HDIM + k * 32, HDIM, lane);
      ar  = WMMA_F16(a, b, ar);
      b = load_bt_rm(dWhh16 + (1 * HDIM + j * 16) * HDIM + k * 32, HDIM, lane);
      az  = WMMA_F16(a, b, az);
      b = load_bt_rm(dWhh16 + (2 * HDIM + j * 16) * HDIM + k * 32, HDIM, lane);
      anh = WMMA_F16(a, b, anh);
    }
    const float w0r = dWih32[(0 * HDIM + col) * KW_DEC];
    const float w0z = dWih32[(1 * HDIM + col) * KW_DEC];
    const float w0n = dWih32[(2 * HDIM + col) * KW_DEC];
    const float br_ = bih[col] + bhh[col];
    const float bz_ = bih[HDIM + col] + bhh[HDIM + col];
    const float bxn = bih[2 * HDIM + col];
    const float bhn = bhh[2 * HDIM + col];
#pragma unroll
    for (int r = 0; r < 8; ++r) {
      const int m = mb + r;
      const float tg = tgt[m];
      const float gr = ar[r] + br_ + tg * w0r;
      const float gz = az[r] + bz_ + tg * w0z;
      const float xn = anx[r] + bxn + tg * w0n;
      const float hn = anh[r] + bhn;
      const float rg = sigf(gr), zg = sigf(gz);
      const float ng = tanhf(xn + rg * hn);
      const float hold = h32[m * HDIM + col];
      const float h2 = (1.f - zg) * ng + zg * hold;
      h2s[m * HDIM + col] = (_Float16)h2;
      h32[m * HDIM + col] = h2;              // in-place: same-thread RMW only
      hout16[m * HDIM + col] = (_Float16)h2; // double-buffered for next step
    }
  }
  __syncthreads();

  // pred[b] = [h2, weighted, tgt] . fc_out_W + fc_out_b ; 2 rows per wave
#pragma unroll 1
  for (int rr = 0; rr < 2; ++rr) {
    const int b = wave + rr * 32;
    float p = 0.f;
    for (int k = lane; k < 769; k += 32) {
      float v;
      if (k < HDIM)     v = (float)h2s[b * HDIM + k];
      else if (k < 768) v = w32[b * H2 + (k - HDIM)];
      else              v = tgt[b];
      p += v * fcW[k];
    }
#pragma unroll
    for (int off = 16; off; off >>= 1) p += __shfl_xor(p, off);
    if (lane == 0) outp[b] = p + fcb[0];
  }
}

// ---------------------------------------------------------------------------
// Host orchestration
// ---------------------------------------------------------------------------
extern "C" void kernel_launch(void* const* d_in, const int* in_sizes, int n_in,
                              void* d_out, int out_size, void* d_ws, size_t ws_size,
                              hipStream_t stream) {
  (void)in_sizes; (void)n_in; (void)out_size; (void)ws_size;

  const float* x        = (const float*)d_in[0];
  const float* tgt      = (const float*)d_in[1];
  const float* WihF     = (const float*)d_in[2];
  const float* WhhF     = (const float*)d_in[3];
  const float* bihF     = (const float*)d_in[4];
  const float* bhhF     = (const float*)d_in[5];
  const float* WihB     = (const float*)d_in[6];
  const float* WhhB     = (const float*)d_in[7];
  const float* bihB     = (const float*)d_in[8];
  const float* bhhB     = (const float*)d_in[9];
  const float* encFcW   = (const float*)d_in[10];
  const float* encFcB   = (const float*)d_in[11];
  const float* attnW    = (const float*)d_in[12];
  const float* attnB    = (const float*)d_in[13];
  const float* vW       = (const float*)d_in[14];
  const float* dWih     = (const float*)d_in[15];
  const float* dWhh     = (const float*)d_in[16];
  const float* dbih     = (const float*)d_in[17];
  const float* dbhh     = (const float*)d_in[18];
  const float* fcOutW   = (const float*)d_in[19];
  const float* fcOutB   = (const float*)d_in[20];
  float* out = (float*)d_out;

  // Workspace carve
  char* base = (char*)d_ws;
  size_t off = 0;
  auto carve = [&](size_t bytes) -> void* {
    void* p = base + off;
    off = (off + bytes + 255) & ~(size_t)255;
    return p;
  };
  _Float16* enc16   = (_Float16*)carve((size_t)BSZ * SRC_LEN * H2 * 2);   // 134 MB
  _Float16* E16     = (_Float16*)carve((size_t)BSZ * SRC_LEN * HDIM * 2); //  67 MB
  _Float16* WhhF16  = (_Float16*)carve((size_t)G3 * HDIM * 2);
  _Float16* WhhB16  = (_Float16*)carve((size_t)G3 * HDIM * 2);
  _Float16* attnW16 = (_Float16*)carve((size_t)HDIM * G3 * 2);
  _Float16* fcW16   = (_Float16*)carve((size_t)HDIM * H2 * 2);
  _Float16* dWihP16 = (_Float16*)carve((size_t)G3 * H2 * 2);   // repacked, ld=512
  _Float16* dWhh16  = (_Float16*)carve((size_t)G3 * HDIM * 2);
  float*    h32     = (float*)carve((size_t)BSZ * HDIM * 4);
  _Float16* h16a    = (_Float16*)carve((size_t)BSZ * HDIM * 2);
  _Float16* h16b    = (_Float16*)carve((size_t)BSZ * HDIM * 2);
  float*    q32     = (float*)carve((size_t)BSZ * HDIM * 4);
  float*    sc      = (float*)carve((size_t)BSZ * SRC_LEN * 4);
  float*    w32     = (float*)carve((size_t)BSZ * H2 * 4);
  _Float16* w16     = (_Float16*)carve((size_t)BSZ * H2 * 2);

  // Weight conversions to f16
  auto cvt = [&](const float* s, _Float16* d, int n) {
    k_cvt<<<(n + 255) / 256, 256, 0, stream>>>(s, d, n);
  };
  cvt(WhhF,   WhhF16,  G3 * HDIM);
  cvt(WhhB,   WhhB16,  G3 * HDIM);
  cvt(attnW,  attnW16, HDIM * G3);
  cvt(encFcW, fcW16,   HDIM * H2);
  cvt(dWhh,   dWhh16,  G3 * HDIM);
  k_cvt_dwih<<<(G3 * H2) / 256, 256, 0, stream>>>(dWih, dWihP16);

  // Encoder (2 persistent workgroups: forward + backward)
  k_encoder<<<2, 1024, 0, stream>>>(x, WihF, bihF, bhhF, WihB, bihB, bhhB,
                                    WhhF16, WhhB16, enc16);

  // Step-invariant attention precompute E = enc @ attn_We^T
  // One block per 16-row M strip; TDM stages A into LDS.
  k_gemm_E<<<BSZ * SRC_LEN / 16, 512, 0, stream>>>(enc16, attnW16, E16);

  // Decoder initial hidden
  k_enc_hidden<<<1, 1024, 0, stream>>>(enc16, fcW16, encFcB, h32, h16a);

  // Decoder loop (hidden state f16 double-buffered)
  const int score_rows = BSZ * SRC_LEN;  // 131072
  for (int t = 0; t < TRG_LEN; ++t) {
    const _Float16* hin = (t & 1) ? h16b : h16a;
    _Float16* hout      = (t & 1) ? h16a : h16b;
    k_dec_q<<<1, 1024, 0, stream>>>(hin, attnW16, attnB, q32);
    k_scores<<<score_rows / 8, 256, 0, stream>>>(E16, q32, vW, sc);
    k_attn_apply<<<BSZ, 1024, 0, stream>>>(sc, enc16, w32, w16);
    k_dec_gru<<<1, 1024, 0, stream>>>(tgt + (size_t)t * BSZ, w32, w16, h32,
                                      hin, hout, dWihP16, dWhh16, dWih, dbih,
                                      dbhh, fcOutW, fcOutB, out + (size_t)t * BSZ);
  }
}
